// Block_12687333392720
// MI455X (gfx1250) — compile-verified, matrix operand order checked by reading, not run on hardware
//
#include <hip/hip_runtime.h>
#include <hip/hip_bf16.h>
#include <math.h>

typedef __bf16 bf16;
typedef __attribute__((ext_vector_type(16))) __bf16 v16bf;
typedef __attribute__((ext_vector_type(8)))  float  v8f;
typedef __attribute__((ext_vector_type(4)))  unsigned int v4u;
typedef __attribute__((ext_vector_type(8)))  int    v8i;
typedef __attribute__((ext_vector_type(4)))  int    v4i;

union Frag {
    v16bf v;
    uint4 q[2];
    bf16  e[16];
};

__device__ __forceinline__ v8f wmma_bf16(v16bf a, v16bf b, v8f c) {
    // D(f32 16x16) = A(bf16 16x32) * B(bf16 32x16) + C
    return __builtin_amdgcn_wmma_f32_16x16x32_bf16(
        /*neg_a=*/false, a, /*neg_b=*/false, b,
        /*c_mod=*/(short)0, c, /*reuse_a=*/false, /*reuse_b=*/false);
}

// ---------------------------------------------------------------------------
// f32 -> bf16 cast (weights / activations)
// ---------------------------------------------------------------------------
__global__ void cast_bf16_kernel(const float* __restrict__ s,
                                 bf16* __restrict__ d, int n) {
    for (int i = blockIdx.x * blockDim.x + threadIdx.x; i < n;
         i += gridDim.x * blockDim.x)
        d[i] = (bf16)s[i];
}

// ---------------------------------------------------------------------------
// LayerNorm: one 256-thread block per row, bf16 output
// ---------------------------------------------------------------------------
__global__ __launch_bounds__(256) void ln_kernel(const float* __restrict__ x,
                                                 const float* __restrict__ w,
                                                 const float* __restrict__ b,
                                                 bf16* __restrict__ out,
                                                 int Cdim) {
    __shared__ float red[8];
    const int row = blockIdx.x;
    const float* xr = x + (size_t)row * Cdim;

    float acc = 0.f;
    for (int i = threadIdx.x; i < Cdim; i += 256) acc += xr[i];
    #pragma unroll
    for (int o = 16; o; o >>= 1) acc += __shfl_xor(acc, o, 32);
    if ((threadIdx.x & 31) == 0) red[threadIdx.x >> 5] = acc;
    __syncthreads();
    float mu = 0.f;
    #pragma unroll
    for (int i = 0; i < 8; ++i) mu += red[i];
    mu /= (float)Cdim;

    float vacc = 0.f;
    for (int i = threadIdx.x; i < Cdim; i += 256) {
        float dlt = xr[i] - mu;
        vacc += dlt * dlt;
    }
    #pragma unroll
    for (int o = 16; o; o >>= 1) vacc += __shfl_xor(vacc, o, 32);
    __syncthreads();
    if ((threadIdx.x & 31) == 0) red[threadIdx.x >> 5] = vacc;
    __syncthreads();
    float var = 0.f;
    #pragma unroll
    for (int i = 0; i < 8; ++i) var += red[i];
    var /= (float)Cdim;
    const float rs = rsqrtf(var + 1e-5f);

    bf16* orow = out + (size_t)row * Cdim;
    for (int i = threadIdx.x; i < Cdim; i += 256)
        orow[i] = (bf16)((xr[i] - mu) * rs * w[i] + b[i]);
}

// ---------------------------------------------------------------------------
// WMMA GEMM: D = A[MxK] * B[KxN] + bias  (A,B bf16 row-major)
// Block tile 128(M) x 64(N), K-step 32. 8 waves, each wave: 16 x 64.
// A tile staged via Tensor Data Mover (TDM) with hardware LDS padding
// (pad_interval=16 DWORDs, pad_amount=4 DWORDs -> element stride 40).
// EPI 0: bf16 out;  EPI 1: f32 out = v + resid;  EPI 2: bf16 out = gelu(v)
// ---------------------------------------------------------------------------
template <int EPI>
__global__ __launch_bounds__(256) void gemm_wmma_kernel(
    const bf16* __restrict__ A, const bf16* __restrict__ Bw,
    const float* __restrict__ bias, const float* __restrict__ resid,
    float* __restrict__ outF, bf16* __restrict__ outH,
    int M, int N, int K) {
    constexpr int LDA = 40;   // element stride (80 B = 5*16 B, keeps b128 aligned)
    __shared__ bf16 sA[128 * LDA];       // 128 rows x 32 K (TDM-padded)
    __shared__ bf16 sB[64 * LDA];        // 64 N-cols x 32 K (transposed, padded)

    const int tid  = threadIdx.x;
    const int lane = tid & 31;
    const int wave = tid >> 5;
    const int hi   = lane >> 4;
    const int ln   = lane & 15;
    const int m0 = blockIdx.y * 128;
    const int n0 = blockIdx.x * 64;

    const unsigned int ldsA = (unsigned int)(size_t)&sA[0];

    v8f acc[4] = {v8f{}, v8f{}, v8f{}, v8f{}};

    for (int k0 = 0; k0 < K; k0 += 32) {
        // ---- stage A via TDM: 128x32 bf16 tile, HW pad to stride 40 ----
        if (wave == 0) {
            unsigned long long ga =
                (unsigned long long)(size_t)(A + (size_t)m0 * K + k0);
            v4u g0;
            g0[0] = 1u;                                     // count=1
            g0[1] = ldsA;                                   // lds_addr
            g0[2] = (unsigned int)ga;                       // global_addr lo
            g0[3] = (unsigned int)((ga >> 32) & 0x01FFFFFFu)
                    | (2u << 30);                           // type=2 (image)
            v8i g1;
            g1[0] = (1 << 16)      // data_size = 2 bytes
                  | (1 << 20)      // pad_enable
                  | (3 << 22)      // pad_interval: 16 DWORDs (one 32-elem row)
                  | (3 << 25);     // pad_amount: 4 DWORDs (8 elems) -> LDA 40
            g1[1] = (int)((unsigned)(K & 0xFFFF) << 16);    // tensor_dim0 lo16
            g1[2] = (int)(((unsigned)(K >> 16) & 0xFFFFu)   // tensor_dim0 hi16
                  | ((unsigned)(M & 0xFFFF) << 16));        // tensor_dim1 lo16
            g1[3] = (int)(((unsigned)(M >> 16) & 0xFFFFu)   // tensor_dim1 hi16
                  | (32u << 16));                           // tile_dim0 = 32
            g1[4] = 128;                                    // tile_dim1=128, d2=0
            g1[5] = K;                                      // dim0_stride lo32
            g1[6] = 0;
            g1[7] = 0;
            v4i g2 = {0, 0, 0, 0};
            v4i g3 = {0, 0, 0, 0};
            v8i g4 = {0, 0, 0, 0, 0, 0, 0, 0};
            __builtin_amdgcn_tensor_load_to_lds(g0, g1, g2, g3, g4, 0);
            __builtin_amdgcn_s_wait_tensorcnt(0);
        }
        // ---- stage B transposed: [32 K][64 N] -> sB[n][k] ----
        {
            const int r  = tid >> 3;
            const int c8 = (tid & 7) * 8;
            uint4 d = *(const uint4*)(Bw + (size_t)(k0 + r) * N + n0 + c8);
            bf16 tmp[8];
            *(uint4*)tmp = d;
            #pragma unroll
            for (int i = 0; i < 8; ++i) sB[(c8 + i) * LDA + r] = tmp[i];
        }
        if (k0 + 32 < K) {
            __builtin_prefetch(Bw + (size_t)(k0 + 32 + (tid >> 3)) * N + n0, 0, 0);
        }
        __syncthreads();

        // ---- A fragment: wave's 16 rows, this lane's row = ln ----
        Frag fa;
        {
            const bf16* p = sA + (wave * 16 + ln) * LDA + hi * 8;
            fa.q[0] = *(const uint4*)p;
            fa.q[1] = *(const uint4*)(p + 16);
        }
        // ---- 4 B fragments / 4 WMMAs ----
        #pragma unroll
        for (int nb = 0; nb < 4; ++nb) {
            Frag fb;
            const bf16* p = sB + (nb * 16 + ln) * LDA + hi * 8;
            fb.q[0] = *(const uint4*)p;
            fb.q[1] = *(const uint4*)(p + 16);
            acc[nb] = wmma_bf16(fa.v, fb.v, acc[nb]);
        }
        __syncthreads();
    }

    // ---- epilogue: C-layout element (m,n): m = 16*wave + 8*hi + r, n = nb*16+ln
    #pragma unroll
    for (int nb = 0; nb < 4; ++nb) {
        #pragma unroll
        for (int r = 0; r < 8; ++r) {
            const int m = m0 + wave * 16 + hi * 8 + r;
            const int n = n0 + nb * 16 + ln;
            float v = acc[nb][r] + bias[n];
            if (EPI == 1) {
                outF[(size_t)m * N + n] = v + resid[(size_t)m * N + n];
            } else if (EPI == 2) {
                v = 0.5f * v * (1.f + erff(v * 0.70710678118f));
                outH[(size_t)m * N + n] = (bf16)v;
            } else {
                outH[(size_t)m * N + n] = (bf16)v;
            }
        }
    }
}

// ---------------------------------------------------------------------------
// Flash attention: one wave per (b, h, 16-query tile); stream 32-key tiles.
// qkv: bf16 [B, T, 3C] (q at 0, k at +C, v at +2C); out: bf16 [B, T, C]
// ---------------------------------------------------------------------------
__global__ __launch_bounds__(32) void flash_attn_kernel(
    const bf16* __restrict__ qkv, bf16* __restrict__ out) {
    constexpr int Cc = 768, Tt = 2048, HD = 64;
    const int qt   = blockIdx.x;     // query tile (16 rows)
    const int h    = blockIdx.y;
    const int b    = blockIdx.z;
    const int lane = threadIdx.x;
    const int hi   = lane >> 4;
    const int ln   = lane & 15;

    __shared__ bf16 sP[16 * 32];     // probabilities, restaged for A-layout
    __shared__ bf16 sV[32 * 64];     // V tile row-major

    const size_t rstride = 3 * Cc;
    const bf16* qbase = qkv + (size_t)b * Tt * rstride + h * HD;

    // Q fragments: 16x64 -> two 16x32 A-frags
    v16bf qf[2];
    #pragma unroll
    for (int kd = 0; kd < 2; ++kd) {
        Frag f;
        const bf16* p =
            qbase + (size_t)(qt * 16 + ln) * rstride + kd * 32 + hi * 8;
        f.q[0] = *(const uint4*)p;
        f.q[1] = *(const uint4*)(p + 16);
        qf[kd] = f.v;
    }

    v8f o[4] = {v8f{}, v8f{}, v8f{}, v8f{}};
    float rmax[8], rsum[8];
    #pragma unroll
    for (int r = 0; r < 8; ++r) { rmax[r] = -1e30f; rsum[r] = 0.f; }

    const int nkv = (qt * 16 + 15) / 32 + 1;   // 32-key tiles (causal)
    const float scale = 0.125f;                // 1/sqrt(64)

    for (int kt = 0; kt < nkv; ++kt) {
        // ---- S = Q K^T for 32 keys: two 16x16 C-frags, 4 WMMAs ----
        v8f s[2] = {v8f{}, v8f{}};
        #pragma unroll
        for (int nb = 0; nb < 2; ++nb) {
            const int key = kt * 32 + nb * 16 + ln;
            const bf16* kbase =
                qkv + ((size_t)b * Tt + key) * rstride + Cc + h * HD;
            #pragma unroll
            for (int kd = 0; kd < 2; ++kd) {
                Frag f;
                const bf16* p = kbase + kd * 32 + hi * 8;
                f.q[0] = *(const uint4*)p;
                f.q[1] = *(const uint4*)(p + 16);
                s[nb] = wmma_bf16(qf[kd], f.v, s[nb]);
            }
        }

        // ---- scale + causal mask + online softmax ----
        float pv[2][8];
        #pragma unroll
        for (int nb = 0; nb < 2; ++nb)
            #pragma unroll
            for (int r = 0; r < 8; ++r) {
                const int qq = qt * 16 + hi * 8 + r;
                const int kk = kt * 32 + nb * 16 + ln;
                const float v = s[nb][r] * scale;
                pv[nb][r] = (kk <= qq) ? v : -1e30f;
            }
        #pragma unroll
        for (int r = 0; r < 8; ++r) {
            float mx = fmaxf(pv[0][r], pv[1][r]);
            #pragma unroll
            for (int oo = 1; oo < 16; oo <<= 1)
                mx = fmaxf(mx, __shfl_xor(mx, oo, 32));
            const float nm = fmaxf(rmax[r], mx);
            const float corr = __expf(rmax[r] - nm);
            rmax[r] = nm;
            float rs = 0.f;
            #pragma unroll
            for (int nb = 0; nb < 2; ++nb) {
                const float e = __expf(pv[nb][r] - nm);
                pv[nb][r] = e;
                rs += e;
            }
            #pragma unroll
            for (int oo = 1; oo < 16; oo <<= 1)
                rs += __shfl_xor(rs, oo, 32);
            rsum[r] = rsum[r] * corr + rs;
            #pragma unroll
            for (int nb = 0; nb < 4; ++nb) o[nb][r] *= corr;
        }

        // ---- restage P (C-layout -> LDS row-major 16x32, bf16) ----
        #pragma unroll
        for (int nb = 0; nb < 2; ++nb)
            #pragma unroll
            for (int r = 0; r < 8; ++r)
                sP[(hi * 8 + r) * 32 + nb * 16 + ln] = (bf16)pv[nb][r];

        // ---- stage V tile 32x64 (one row per lane, b128s) ----
        {
            const int key = kt * 32 + lane;
            const bf16* vbase =
                qkv + ((size_t)b * Tt + key) * rstride + 2 * Cc + h * HD;
            #pragma unroll
            for (int c = 0; c < 64; c += 8)
                *(uint4*)(sV + lane * 64 + c) = *(const uint4*)(vbase + c);
        }
        __syncthreads();   // single-wave WG: S_NOP, but compiler mem-fence

        // ---- O += P(16x32) @ V(32x64): A-frag + 4 B-frags, 4 WMMAs ----
        Frag fp;
        {
            const bf16* p = sP + ln * 32 + hi * 8;
            fp.q[0] = *(const uint4*)p;
            fp.q[1] = *(const uint4*)(p + 16);
        }
        #pragma unroll
        for (int nb = 0; nb < 4; ++nb) {
            Frag fv;
            #pragma unroll
            for (int i = 0; i < 8; ++i)
                fv.e[i] = sV[(hi * 8 + i) * 64 + nb * 16 + ln];
            #pragma unroll
            for (int i = 0; i < 8; ++i)
                fv.e[8 + i] = sV[(hi * 8 + 16 + i) * 64 + nb * 16 + ln];
            o[nb] = wmma_bf16(fp.v, fv.v, o[nb]);
        }
        __syncthreads();
    }

    // ---- normalize + write out [B,T,C] slice for this head ----
    bf16* ob = out + ((size_t)b * Tt + qt * 16) * Cc + h * HD;
    #pragma unroll
    for (int nb = 0; nb < 4; ++nb)
        #pragma unroll
        for (int r = 0; r < 8; ++r) {
            const float v = o[nb][r] / rsum[r];
            ob[(size_t)(hi * 8 + r) * Cc + nb * 16 + ln] = (bf16)v;
        }
}

// ---------------------------------------------------------------------------
extern "C" void kernel_launch(void* const* d_in, const int* in_sizes, int n_in,
                              void* d_out, int out_size, void* d_ws,
                              size_t ws_size, hipStream_t stream) {
    (void)in_sizes; (void)n_in; (void)out_size; (void)ws_size;
    constexpr int Bb = 4, Tt = 2048, Cc = 768, NH = 12;
    constexpr int M = Bb * Tt;            // 8192

    const float* x       = (const float*)d_in[0];
    const float* ln1_w   = (const float*)d_in[1];
    const float* ln1_b   = (const float*)d_in[2];
    const float* W_qkv   = (const float*)d_in[3];
    const float* b_qkv   = (const float*)d_in[4];
    const float* W_ap    = (const float*)d_in[5];
    const float* b_ap    = (const float*)d_in[6];
    const float* ln2_w   = (const float*)d_in[7];
    const float* ln2_b   = (const float*)d_in[8];
    const float* W_fc    = (const float*)d_in[9];
    const float* b_fc    = (const float*)d_in[10];
    const float* W_mp    = (const float*)d_in[11];
    const float* b_mp    = (const float*)d_in[12];

    // ---- workspace carve-up (256-B aligned) ----
    char* w = (char*)d_ws;
    auto carve = [&](size_t bytes) {
        char* p = w;
        w += (bytes + 255) & ~(size_t)255;
        return p;
    };
    bf16*  ln1o  = (bf16*)carve((size_t)M * Cc * 2);
    bf16*  Wqkvh = (bf16*)carve((size_t)Cc * 3 * Cc * 2);
    bf16*  qkv   = (bf16*)carve((size_t)M * 3 * Cc * 2);
    bf16*  attn  = (bf16*)carve((size_t)M * Cc * 2);
    bf16*  Waph  = (bf16*)carve((size_t)Cc * Cc * 2);
    float* xmid  = (float*)carve((size_t)M * Cc * 4);
    bf16*  ln2o  = (bf16*)carve((size_t)M * Cc * 2);
    bf16*  Wfch  = (bf16*)carve((size_t)Cc * 4 * Cc * 2);
    bf16*  h2    = (bf16*)carve((size_t)M * 4 * Cc * 2);
    bf16*  Wmph  = (bf16*)carve((size_t)4 * Cc * Cc * 2);

    // ---- weight casts ----
    cast_bf16_kernel<<<512, 256, 0, stream>>>(W_qkv, Wqkvh, Cc * 3 * Cc);
    cast_bf16_kernel<<<512, 256, 0, stream>>>(W_ap,  Waph,  Cc * Cc);
    cast_bf16_kernel<<<512, 256, 0, stream>>>(W_fc,  Wfch,  Cc * 4 * Cc);
    cast_bf16_kernel<<<512, 256, 0, stream>>>(W_mp,  Wmph,  4 * Cc * Cc);

    // ---- LN1 ----
    ln_kernel<<<M, 256, 0, stream>>>(x, ln1_w, ln1_b, ln1o, Cc);

    // ---- QKV GEMM: [8192,768] x [768,2304] -> bf16 ----
    gemm_wmma_kernel<0><<<dim3(3 * Cc / 64, M / 128), 256, 0, stream>>>(
        ln1o, Wqkvh, b_qkv, nullptr, nullptr, qkv, M, 3 * Cc, Cc);

    // ---- fused causal attention ----
    flash_attn_kernel<<<dim3(Tt / 16, NH, Bb), 32, 0, stream>>>(qkv, attn);

    // ---- attention projection + residual -> f32 x_mid ----
    gemm_wmma_kernel<1><<<dim3(Cc / 64, M / 128), 256, 0, stream>>>(
        attn, Waph, b_ap, x, xmid, nullptr, M, Cc, Cc);

    // ---- LN2 ----
    ln_kernel<<<M, 256, 0, stream>>>(xmid, ln2_w, ln2_b, ln2o, Cc);

    // ---- FC + exact GELU: [8192,768] x [768,3072] -> bf16 ----
    gemm_wmma_kernel<2><<<dim3(4 * Cc / 64, M / 128), 256, 0, stream>>>(
        ln2o, Wfch, b_fc, nullptr, nullptr, h2, M, 4 * Cc, Cc);

    // ---- MLP projection + residual -> f32 d_out ----
    gemm_wmma_kernel<1><<<dim3(Cc / 64, M / 128), 256, 0, stream>>>(
        h2, Wmph, b_mp, xmid, (float*)d_out, nullptr, M, Cc, 4 * Cc);
}
